// SimpleLSTMDecoder_46256797778273
// MI455X (gfx1250) — compile-verified, matrix operand order checked by reading, not run on hardware
//
#include <hip/hip_runtime.h>
#include <hip/hip_bf16.h>

// ---------------------------------------------------------------------------
// Problem constants (match reference)
// ---------------------------------------------------------------------------
constexpr int V  = 32000;
constexpr int E  = 512;
constexpr int H  = 1024;
constexpr int B  = 32;
constexpr int T  = 256;
constexpr int FH = 4 * H;          // 4096 gate columns
constexpr int M  = T * B;          // 8192 time*batch rows
constexpr int SCAN_WGS = 8;        // persistent workgroups for the scan (64 waves)

typedef __attribute__((ext_vector_type(16))) __bf16 v16bf;
typedef __attribute__((ext_vector_type(8)))  __bf16 v8bf;
typedef __attribute__((ext_vector_type(8)))  float  v8f;

// ---------------------------------------------------------------------------
// WMMA helpers
// ---------------------------------------------------------------------------
__device__ __forceinline__ v8f wmma_bf16(v16bf a, v16bf b, v8f c) {
  // D = A(16x32) * B(32x16) + C, fp32 accumulate
  return __builtin_amdgcn_wmma_f32_16x16x32_bf16(
      /*neg_a=*/false, a, /*neg_b=*/false, b,
      /*c_mod=*/(short)0, c, /*reuse_a=*/false, /*reuse_b=*/false);
}

// Load one 16x32 bf16 operand tile in the CDNA5 A/B register layout.
// ISA layout (16-bit A, 16x32): lane&15 selects row M; lanes 0-15 hold
// K={0..7,16..23}, lanes 16-31 hold K={8..15,24..31} -> two 16B chunks/lane.
__device__ __forceinline__ v16bf load_tile16x32(const __bf16* __restrict__ base,
                                                size_t row_stride,
                                                int row0, int k0, int lane) {
  const int half = lane >> 4;
  const int m    = lane & 15;
  const __bf16* p = base + (size_t)(row0 + m) * row_stride + (size_t)(k0 + 8 * half);
  v8bf lo = *(const v8bf*)p;            // K = k0 + 8*half + {0..7}
  v8bf hi = *(const v8bf*)(p + 16);     // K = k0 + 16 + 8*half + {0..7}
  v16bf r;
#pragma unroll
  for (int i = 0; i < 8; ++i) { r[i] = lo[i]; r[i + 8] = hi[i]; }
  return r;
}

__device__ __forceinline__ float sigf(float x) {
  return 1.0f / (1.0f + __expf(-x));
}

__device__ __forceinline__ v8f vzero8() {
  v8f z = {0.f, 0.f, 0.f, 0.f, 0.f, 0.f, 0.f, 0.f};
  return z;
}

// ---------------------------------------------------------------------------
// Small utility kernels
// ---------------------------------------------------------------------------
__global__ void zero_u32(unsigned* p, int n) {
  int i = threadIdx.x;
  if (i < n) p[i] = 0u;
}

__global__ void f32_to_bf16(const float* __restrict__ src,
                            __bf16* __restrict__ dst, size_t n) {
  size_t i      = (size_t)blockIdx.x * blockDim.x + threadIdx.x;
  size_t stride = (size_t)gridDim.x * blockDim.x;
  for (; i < n; i += stride) dst[i] = (__bf16)src[i];
}

// xs[m=t*B+b][e] = bf16(emb[target[b][t]][e]),  target is (B, T+1)
__global__ void gather_embed(const int* __restrict__ target,
                             const float* __restrict__ emb,
                             __bf16* __restrict__ Xbf) {
  size_t idx = (size_t)blockIdx.x * blockDim.x + threadIdx.x;
  if (idx >= (size_t)M * E) return;
  int m = (int)(idx / E);
  int e = (int)(idx % E);
  int t = m / B, b = m % B;
  int tok = target[b * (T + 1) + t];
  Xbf[idx] = (__bf16)emb[(size_t)tok * E + e];
}

// Initialize fp32 cell state + bf16 hidden state buffers
__global__ void init_state(const float* __restrict__ h0,
                           const float* __restrict__ c0,
                           __bf16* __restrict__ hbf,
                           float* __restrict__ cbuf, int n) {
  int i = blockIdx.x * blockDim.x + threadIdx.x;
  if (i < n) { hbf[i] = (__bf16)h0[i]; cbuf[i] = c0[i]; }
}

// ---------------------------------------------------------------------------
// Generic C(MxN,f32) = A(MxK,bf16) @ W(NxK,bf16)^T + bias1 + bias2
// 2x2 register blocking: 4 accumulators per wave, 4 operand-tile loads feed
// 4 WMMAs per k-step (1 tile-load per matrix op).
// ---------------------------------------------------------------------------
__global__ __launch_bounds__(256)
void gemm_bf16_nt(const __bf16* __restrict__ A, const __bf16* __restrict__ W,
                  const float* __restrict__ bias1, const float* __restrict__ bias2,
                  float* __restrict__ C, int Mrows, int N, int K) {
  const int lane = threadIdx.x & 31;
  const int wave = threadIdx.x >> 5;
  long job = (long)blockIdx.x * (blockDim.x >> 5) + wave;
  const int  np    = N >> 5;                 // 32-wide N panels
  const long total = (long)(Mrows >> 5) * np;
  if (job >= total) return;
  const int ntp = (int)(job % np);
  const int mtp = (int)(job / np);

  v8f acc00 = vzero8(), acc01 = vzero8(), acc10 = vzero8(), acc11 = vzero8();
  for (int k0 = 0; k0 < K; k0 += 32) {
    v16bf a0 = load_tile16x32(A, (size_t)K, mtp * 32,      k0, lane);
    v16bf a1 = load_tile16x32(A, (size_t)K, mtp * 32 + 16, k0, lane);
    v16bf b0 = load_tile16x32(W, (size_t)K, ntp * 32,      k0, lane);
    v16bf b1 = load_tile16x32(W, (size_t)K, ntp * 32 + 16, k0, lane);
    acc00 = wmma_bf16(a0, b0, acc00);
    acc01 = wmma_bf16(a0, b1, acc01);
    acc10 = wmma_bf16(a1, b0, acc10);
    acc11 = wmma_bf16(a1, b1, acc11);
  }

  v8f* accs[2][2] = {{&acc00, &acc01}, {&acc10, &acc11}};
#pragma unroll
  for (int mi = 0; mi < 2; ++mi) {
#pragma unroll
    for (int ni = 0; ni < 2; ++ni) {
      const int n = ntp * 32 + ni * 16 + (lane & 15);
      float bv = 0.f;
      if (bias1) bv += bias1[n];
      if (bias2) bv += bias2[n];
      const int mbase = mtp * 32 + mi * 16 + 8 * (lane >> 4);
      v8f a = *accs[mi][ni];
#pragma unroll
      for (int r = 0; r < 8; ++r)
        C[(size_t)(mbase + r) * N + n] = a[r] + bv;
    }
  }
}

// ---------------------------------------------------------------------------
// Device-wide sense-reversal barrier (all SCAN_WGS blocks are resident)
// ---------------------------------------------------------------------------
__device__ __forceinline__ void grid_sync(unsigned* bar, unsigned* gen, unsigned nwg) {
  __syncthreads();
  if (threadIdx.x == 0) {
    __threadfence();                              // release my stores
    unsigned my_gen = atomicAdd(gen, 0u);
    unsigned prev   = atomicAdd(bar, 1u);
    if (prev == nwg - 1u) {
      atomicExch(bar, 0u);
      __threadfence();
      atomicAdd(gen, 1u);
    } else {
      while (atomicAdd(gen, 0u) == my_gen) {
        __builtin_amdgcn_s_sleep(2);
      }
    }
    __threadfence();                              // acquire others' stores
  }
  __syncthreads();
}

// ---------------------------------------------------------------------------
// Persistent LSTM scan over T steps for one layer.
//   gates(t) = Gx[t] (precomputed x-path + biases, fp32) + h(t-1) @ Whh^T
// 64 waves (8 WGs x 8): each wave owns one H-tile j and computes BOTH batch
// m-tiles for all 4 gates (8 accumulators). B-tiles of Whh are therefore
// loaded exactly once per step grid-wide (the 8 MB minimum), and h(t-1)
// (64 KB bf16) is staged once per block into LDS so A-tiles come from
// ds_load_b128 instead of 64x-redundant L2 reads.
// fp32 cell state is owner-exclusive; h state double-buffered in bf16 ->
// exactly one grid_sync per step.
// ---------------------------------------------------------------------------
__global__ __launch_bounds__(256)
void lstm_scan(const float* __restrict__ Gx, const __bf16* __restrict__ Whh,
               float* __restrict__ cbuf,
               __bf16* __restrict__ hbuf0, __bf16* __restrict__ hbuf1,
               __bf16* __restrict__ Hout,
               unsigned* bar, unsigned* gen) {
  __shared__ __bf16 hsh[B * H];                    // 64 KB staged h(t-1)
  const int lane = threadIdx.x & 31;
  const int j    = (blockIdx.x * blockDim.x + threadIdx.x) >> 5;  // 0..63 H-tile
  const int half = lane >> 4;
  const int nl   = lane & 15;
  __bf16* hb[2] = {hbuf0, hbuf1};

  // LDS A-tile load in WMMA operand layout
  auto lds_tile = [&](int row0, int k0) -> v16bf {
    const __bf16* p = &hsh[(size_t)(row0 + nl) * H + (size_t)(k0 + 8 * half)];
    v8bf lo = *(const v8bf*)p;
    v8bf hi = *(const v8bf*)(p + 16);
    v16bf r;
#pragma unroll
    for (int i = 0; i < 8; ++i) { r[i] = lo[i]; r[i + 8] = hi[i]; }
    return r;
  };

#pragma unroll 1
  for (int t = 0; t < T; ++t) {
    const __bf16* hprev = hb[t & 1];
    __bf16*       hnext = hb[(t + 1) & 1];

    // cooperative stage: h(t-1) global -> LDS (B*H bf16 = 64 KB)
    {
      const uint4* s = (const uint4*)hprev;
      uint4*       d = (uint4*)hsh;
#pragma unroll 1
      for (int i = threadIdx.x; i < (B * H) / 8; i += 256) d[i] = s[i];
    }
    __syncthreads();

    const float* gx = Gx + (size_t)t * B * FH;
    v8f acc[4][2];
#pragma unroll
    for (int g = 0; g < 4; ++g) {
      const int col = g * H + j * 16 + nl;
#pragma unroll
      for (int mt = 0; mt < 2; ++mt) {
#pragma unroll
        for (int r = 0; r < 8; ++r)
          acc[g][mt][r] = gx[(size_t)(mt * 16 + 8 * half + r) * FH + col];
      }
    }

#pragma unroll 1
    for (int k0 = 0; k0 < H; k0 += 32) {
      v16bf a0 = lds_tile(0,  k0);
      v16bf a1 = lds_tile(16, k0);
#pragma unroll
      for (int g = 0; g < 4; ++g) {
        v16bf b = load_tile16x32(Whh, (size_t)H, g * H + j * 16, k0, lane);
        acc[g][0] = wmma_bf16(a0, b, acc[g][0]);
        acc[g][1] = wmma_bf16(a1, b, acc[g][1]);
      }
    }

    const int col = j * 16 + nl;
#pragma unroll
    for (int mt = 0; mt < 2; ++mt) {
#pragma unroll
      for (int r = 0; r < 8; ++r) {
        const int brow = mt * 16 + 8 * half + r;   // batch row, owner-exclusive
        const float iv = acc[0][mt][r];
        const float fv = acc[1][mt][r];
        const float gv = acc[2][mt][r];
        const float ov = acc[3][mt][r];
        const size_t ci = (size_t)brow * H + col;
        const float cn = sigf(fv) * cbuf[ci] + sigf(iv) * tanhf(gv);
        const float hn = sigf(ov) * tanhf(cn);
        cbuf[ci] = cn;
        const __bf16 hv = (__bf16)hn;
        hnext[ci] = hv;
        Hout[((size_t)t * B + brow) * H + col] = hv;
      }
    }

    grid_sync(bar, gen, SCAN_WGS);
  }
}

// ---------------------------------------------------------------------------
// logits[b][v][t] = hs[t*B+b] . W_out[v] + b_out[v]
// Per-batch GEMM W_out(VxH) x hs_b^T(HxT): the D-tile N dimension is t, the
// contiguous axis of the (B,V,T) output -> coalesced 64B stores per row.
// 2x2 register blocking (2 v-tiles x 2 t-tiles) halves operand traffic.
// ---------------------------------------------------------------------------
__global__ __launch_bounds__(256)
void out_proj(const __bf16* __restrict__ hs, const __bf16* __restrict__ Wout,
              const float* __restrict__ bout, float* __restrict__ out) {
  const int lane = threadIdx.x & 31;
  const int wave = threadIdx.x >> 5;
  long job = (long)blockIdx.x * (blockDim.x >> 5) + wave;
  constexpr int VP = V / 32;      // 1000 v-panels
  constexpr int TP = T / 32;      // 8 t-panels
  const long total = (long)B * VP * TP;
  if (job >= total) return;
  const int tp = (int)(job % TP);
  const int vp = (int)((job / TP) % VP);
  const int b  = (int)(job / ((long)TP * VP));

  const __bf16* hsb = hs + (size_t)b * H;          // row t lives at stride B*H
  v8f acc00 = vzero8(), acc01 = vzero8(), acc10 = vzero8(), acc11 = vzero8();
  for (int k0 = 0; k0 < H; k0 += 32) {
    v16bf a0 = load_tile16x32(Wout, (size_t)H, vp * 32,      k0, lane);
    v16bf a1 = load_tile16x32(Wout, (size_t)H, vp * 32 + 16, k0, lane);
    v16bf t0 = load_tile16x32(hsb, (size_t)B * H, tp * 32,      k0, lane);
    v16bf t1 = load_tile16x32(hsb, (size_t)B * H, tp * 32 + 16, k0, lane);
    acc00 = wmma_bf16(a0, t0, acc00);
    acc01 = wmma_bf16(a0, t1, acc01);
    acc10 = wmma_bf16(a1, t0, acc10);
    acc11 = wmma_bf16(a1, t1, acc11);
  }

  v8f* accs[2][2] = {{&acc00, &acc01}, {&acc10, &acc11}};
#pragma unroll
  for (int vi = 0; vi < 2; ++vi) {
#pragma unroll
    for (int ti = 0; ti < 2; ++ti) {
      const int tcol  = tp * 32 + ti * 16 + (lane & 15);
      const int vbase = vp * 32 + vi * 16 + 8 * (lane >> 4);
      v8f a = *accs[vi][ti];
#pragma unroll
      for (int r = 0; r < 8; ++r) {
        const int v = vbase + r;
        out[(size_t)b * V * T + (size_t)v * T + tcol] = a[r] + bout[v];
      }
    }
  }
}

// ---------------------------------------------------------------------------
// Host launcher
// ---------------------------------------------------------------------------
extern "C" void kernel_launch(void* const* d_in, const int* in_sizes, int n_in,
                              void* d_out, int out_size, void* d_ws, size_t ws_size,
                              hipStream_t stream) {
  const int*   target = (const int*)d_in[0];
  const float* hidden = (const float*)d_in[1];   // (L,B,H)
  const float* cell   = (const float*)d_in[2];   // (L,B,H)
  const float* emb    = (const float*)d_in[4];
  const float* Wih0   = (const float*)d_in[5];
  const float* Whh0   = (const float*)d_in[6];
  const float* bih0   = (const float*)d_in[7];
  const float* bhh0   = (const float*)d_in[8];
  const float* Wih1   = (const float*)d_in[9];
  const float* Whh1   = (const float*)d_in[10];
  const float* bih1   = (const float*)d_in[11];
  const float* bhh1   = (const float*)d_in[12];
  const float* WoutF  = (const float*)d_in[13];
  const float* bout   = (const float*)d_in[14];
  float* out = (float*)d_out;

  // --- workspace carve-up (256B aligned) ---
  char*  w   = (char*)d_ws;
  size_t off = 0;
  auto alloc = [&](size_t bytes) -> void* {
    void* p = w + off;
    off += (bytes + 255) & ~(size_t)255;
    return p;
  };
  unsigned* bars  = (unsigned*)alloc(256);                 // bar0,gen0,bar1,gen1
  __bf16* Xbf     = (__bf16*)alloc((size_t)M * E * 2);
  __bf16* Wih0b   = (__bf16*)alloc((size_t)FH * E * 2);
  __bf16* Whh0b   = (__bf16*)alloc((size_t)FH * H * 2);
  __bf16* Wih1b   = (__bf16*)alloc((size_t)FH * H * 2);
  __bf16* Whh1b   = (__bf16*)alloc((size_t)FH * H * 2);
  __bf16* Woutb   = (__bf16*)alloc((size_t)V * H * 2);
  float*  Gx      = (float*)alloc((size_t)M * FH * 4);     // reused for both layers
  __bf16* h0s     = (__bf16*)alloc((size_t)M * H * 2);
  __bf16* hs      = (__bf16*)alloc((size_t)M * H * 2);
  float*  c0buf   = (float*)alloc((size_t)B * H * 4);
  float*  c1buf   = (float*)alloc((size_t)B * H * 4);
  __bf16* hA0     = (__bf16*)alloc((size_t)B * H * 2);
  __bf16* hB0     = (__bf16*)alloc((size_t)B * H * 2);
  __bf16* hA1     = (__bf16*)alloc((size_t)B * H * 2);
  __bf16* hB1     = (__bf16*)alloc((size_t)B * H * 2);

  // --- prologue: barriers, weight conversion, embedding gather ---
  zero_u32<<<1, 64, 0, stream>>>(bars, 8);
  f32_to_bf16<<<2048, 256, 0, stream>>>(Wih0, Wih0b, (size_t)FH * E);
  f32_to_bf16<<<2048, 256, 0, stream>>>(Whh0, Whh0b, (size_t)FH * H);
  f32_to_bf16<<<2048, 256, 0, stream>>>(Wih1, Wih1b, (size_t)FH * H);
  f32_to_bf16<<<2048, 256, 0, stream>>>(Whh1, Whh1b, (size_t)FH * H);
  f32_to_bf16<<<4096, 256, 0, stream>>>(WoutF, Woutb, (size_t)V * H);
  gather_embed<<<(int)(((size_t)M * E + 255) / 256), 256, 0, stream>>>(target, emb, Xbf);
  init_state<<<(B * H + 255) / 256, 256, 0, stream>>>(hidden, cell, hA0, c0buf, B * H);
  init_state<<<(B * H + 255) / 256, 256, 0, stream>>>(hidden + (size_t)B * H,
                                                      cell + (size_t)B * H, hA1, c1buf, B * H);

  // --- layer 0: parallel x-path GEMM, then sequential scan ---
  {
    const long jobs = (long)(M / 32) * (FH / 32);
    gemm_bf16_nt<<<(int)(jobs / 8), 256, 0, stream>>>(Xbf, Wih0b, bih0, bhh0, Gx, M, FH, E);
  }
  lstm_scan<<<SCAN_WGS, 256, 0, stream>>>(Gx, Whh0b, c0buf, hA0, hB0, h0s, bars + 0, bars + 1);

  // --- layer 1: parallel x-path GEMM from h0s, then sequential scan ---
  {
    const long jobs = (long)(M / 32) * (FH / 32);
    gemm_bf16_nt<<<(int)(jobs / 8), 256, 0, stream>>>(h0s, Wih1b, bih1, bhh1, Gx, M, FH, H);
  }
  lstm_scan<<<SCAN_WGS, 256, 0, stream>>>(Gx, Whh1b, c1buf, hA1, hB1, hs, bars + 2, bars + 3);

  // --- output projection: (B,V,T) logits ---
  {
    const long jobs = (long)B * (V / 32) * (T / 32);
    out_proj<<<(int)(jobs / 8), 256, 0, stream>>>(hs, Woutb, bout, out);
  }
}
